// SimpleMoELayer_59055800320452
// MI455X (gfx1250) — matrix operand order repaired and checked by reading, not compiled, not run on hardware
//
#include <hip/hip_runtime.h>
#include <hip/hip_bf16.h>
#include <stdint.h>

// ---------------- types ----------------
typedef __attribute__((ext_vector_type(16))) __bf16 v16bf;
typedef __attribute__((ext_vector_type(8)))  float  v8f;
typedef __attribute__((ext_vector_type(4)))  float  f32x4;
typedef __attribute__((ext_vector_type(4)))  unsigned int u32x4;
typedef __attribute__((ext_vector_type(2)))  unsigned int u32x2;

#define T_TOK 4096
#define D_DIM 1024
#define F_DIM 4096
#define E_NUM 8
#define ROWS_CAP 9216            // 2*T + per-expert 128-row padding headroom
#define ROW_TILES (ROWS_CAP/128) // 72

union Frag16 { v16bf v; u32x4 u[2]; };

// A fragment: 16x32 bf16, rows m..m+15. ISA layout: lanes0-15 hold K={0..7,16..23},
// lanes16-31 hold K={8..15,24..31}; LDS tile is row-major [128][32].
__device__ inline v16bf load_frag_A(const __bf16* S, int mBase, int lane) {
  const int m  = mBase + (lane & 15);
  const int hi = lane >> 4;
  const u32x4* row = (const u32x4*)(S + m * 32);
  Frag16 f;
  f.u[0] = row[hi];
  f.u[1] = row[hi + 2];
  return f.v;
}
// B fragment: 32x16 bf16, cols n..n+15. LDS tile stored N-major [128 n][32 k]; same K
// layout as A: lanes0-15 -> K {0..7,16..23}, lanes16-31 -> K {8..15,24..31}.
__device__ inline v16bf load_frag_B(const __bf16* S, int nBase, int lane) {
  const int n  = nBase + (lane & 15);
  const int kg = lane >> 4;
  const u32x4* row = (const u32x4*)(S + n * 32);
  Frag16 f;
  f.u[0] = row[kg];
  f.u[1] = row[kg + 2];
  return f.v;
}

// CDNA5 async global->LDS copy (ASYNCcnt-tracked); bypasses VGPRs.
__device__ inline void async_ld_b128(unsigned lds_byte_off, const void* gaddr) {
  asm volatile("global_load_async_to_lds_b128 %0, %1, off"
               :: "v"(lds_byte_off), "v"(gaddr) : "memory");
}
__device__ inline void wait_async0() {
  asm volatile("s_wait_asynccnt 0" ::: "memory");
}

// Stage B tile (32 K x 128 N) fp32 -> bf16 into Bs[n][k]; each thread converts a
// 2K x 4N micro-tile and stores packed bf16 pairs as b32 (v_cvt_pk_bf16_f32 path).
__device__ inline void stage_B(const float* __restrict__ W, int ldw, int k0, int n0,
                               __bf16* __restrict__ Bs, int tid) {
  #pragma unroll
  for (int j = 0; j < 2; ++j) {
    const int idx = tid + j * 256;       // 0..511
    const int nb  = (idx & 31) * 4;      // 0..124 (lane-major over N -> coalesced)
    const int kp  = idx >> 5;            // 0..15  -> k = 2*kp
    const float* g0 = W + (size_t)(k0 + 2 * kp) * ldw + n0 + nb;
    f32x4 a = *(const f32x4*)g0;
    f32x4 b = *(const f32x4*)(g0 + ldw);
    #pragma unroll
    for (int q = 0; q < 4; ++q) {
      alignas(4) __bf16 pr[2] = { (__bf16)a[q], (__bf16)b[q] };
      *(unsigned*)&Bs[(size_t)(nb + q) * 32 + 2 * kp] = *(const unsigned*)pr;
    }
  }
}

#define WMMA_BF16(A, B, C) \
  __builtin_amdgcn_wmma_f32_16x16x32_bf16(false, (A), false, (B), (short)0, (C), false, false)

// ---------------- small kernels ----------------
__global__ __launch_bounds__(256) void zero_f32(float* __restrict__ p, int n) {
  int i = blockIdx.x * blockDim.x + threadIdx.x;
  if (i < n) p[i] = 0.0f;
}

__global__ __launch_bounds__(256) void init_route(int* pair_count, int* cursor,
                                                  float* top1_cnt, float* prob_sums,
                                                  int* row_token, float* row_weight) {
  int i = blockIdx.x * blockDim.x + threadIdx.x;
  if (i < E_NUM) { pair_count[i] = 0; cursor[i] = 0; top1_cnt[i] = 0.f; prob_sums[i] = 0.f; }
  if (i < ROWS_CAP) { row_token[i] = 0; row_weight[i] = 0.f; }
}

__global__ __launch_bounds__(256) void cvt_bf16(const float* __restrict__ src,
                                                __bf16* __restrict__ dst, int n4) {
  int i = blockIdx.x * blockDim.x + threadIdx.x;
  if (i >= n4) return;
  f32x4 v = ((const f32x4*)src)[i];
  alignas(8) __bf16 o[4] = {(__bf16)v.x, (__bf16)v.y, (__bf16)v.z, (__bf16)v.w};
  *(u32x2*)(dst + (size_t)i * 4) = *(const u32x2*)o;
}

// Gate: logits = x@Wg + bg, top-2 softmax, aux-loss statistics.
__global__ __launch_bounds__(256) void gate_kernel(
    const float* __restrict__ x, const float* __restrict__ Wg, const float* __restrict__ bg,
    int* pair_count, float* top1_cnt, float* prob_sums,
    int* __restrict__ tok_e, float* __restrict__ tok_p) {
  int t = blockIdx.x * blockDim.x + threadIdx.x;
  if (t >= T_TOK) return;
  float l[E_NUM];
  #pragma unroll
  for (int e = 0; e < E_NUM; ++e) l[e] = bg[e];
  const float* xr = x + (size_t)t * D_DIM;
  for (int d = 0; d < D_DIM; ++d) {
    float xv = xr[d];
    #pragma unroll
    for (int e = 0; e < E_NUM; ++e) l[e] += xv * Wg[d * E_NUM + e];
  }
  int i0 = 0;
  #pragma unroll
  for (int e = 1; e < E_NUM; ++e) if (l[e] > l[i0]) i0 = e;
  int i1 = (i0 == 0) ? 1 : 0;
  #pragma unroll
  for (int e = 0; e < E_NUM; ++e) if (e != i0 && l[e] > l[i1]) i1 = e;
  float p0 = 1.f / (1.f + __expf(l[i1] - l[i0]));  // softmax over the top-2 values
  float p1 = 1.f - p0;
  // full softmax (for aux loss mean)
  float m = l[i0], denom = 0.f, pr[E_NUM];
  #pragma unroll
  for (int e = 0; e < E_NUM; ++e) { pr[e] = __expf(l[e] - m); denom += pr[e]; }
  float inv = 1.f / denom;
  #pragma unroll
  for (int e = 0; e < E_NUM; ++e) atomicAdd(&prob_sums[e], pr[e] * inv);
  atomicAdd(&top1_cnt[i0], 1.f);
  atomicAdd(&pair_count[i0], 1);
  atomicAdd(&pair_count[i1], 1);
  tok_e[2 * t] = i0; tok_e[2 * t + 1] = i1;
  tok_p[2 * t] = p0; tok_p[2 * t + 1] = p1;
}

// Prefix sums (padded to 128-row tiles per expert) + aux loss scalar.
__global__ void offsets_kernel(const int* pair_count, const float* top1_cnt,
                               const float* prob_sums, int* padded_off, float* aux_out) {
  if (blockIdx.x == 0 && threadIdx.x == 0) {
    int off = 0;
    padded_off[0] = 0;
    for (int e = 0; e < E_NUM; ++e) {
      off += (pair_count[e] + 127) & ~127;
      padded_off[e + 1] = off;
    }
    float aux = 0.f;
    for (int e = 0; e < E_NUM; ++e)
      aux += (top1_cnt[e] / ((float)T_TOK + 1e-8f)) * (prob_sums[e] / (float)T_TOK);
    *aux_out = aux * (float)E_NUM;
  }
}

__global__ __launch_bounds__(256) void scatter_kernel(
    const int* __restrict__ tok_e, const float* __restrict__ tok_p,
    const int* __restrict__ padded_off, int* cursor,
    int* __restrict__ row_token, float* __restrict__ row_weight) {
  int t = blockIdx.x * blockDim.x + threadIdx.x;
  if (t >= T_TOK) return;
  #pragma unroll
  for (int k = 0; k < 2; ++k) {
    int e = tok_e[2 * t + k];
    int pos = atomicAdd(&cursor[e], 1);
    int r = padded_off[e] + pos;
    row_token[r] = t;
    row_weight[r] = tok_p[2 * t + k];
  }
}

// ---------------- expert GEMMs (bf16 WMMA, f32 accumulate) ----------------
// FFN1: h[row, F] = relu(gather(x)[row, D] @ W1[e] + b1[e]) -> bf16
__global__ __launch_bounds__(256) void moe_ffn1(
    const __bf16* __restrict__ xb, const float* __restrict__ W1,
    const float* __restrict__ b1, const int* __restrict__ row_token,
    const int* __restrict__ padded_off, __bf16* __restrict__ h_buf) {
  __shared__ alignas(16) __bf16 As[2][128 * 32];
  __shared__ alignas(16) __bf16 Bs[2][128 * 32];
  const int m0 = blockIdx.y * 128;
  if (m0 >= padded_off[E_NUM]) return;
  int e = 0;
  while (e < E_NUM - 1 && m0 >= padded_off[e + 1]) ++e;
  const int n0 = blockIdx.x * 128;
  const float* W = W1 + (size_t)e * D_DIM * F_DIM;
  const int tid = threadIdx.x, lane = tid & 31, wid = tid >> 5;
  const int wm = (wid & 3) * 32, wn = (wid >> 2) * 64;
  const int r0 = tid >> 2, c0 = tid & 3;
  const __bf16* gA0 = xb + (size_t)row_token[m0 + r0] * D_DIM + c0 * 8;
  const __bf16* gA1 = xb + (size_t)row_token[m0 + 64 + r0] * D_DIM + c0 * 8;
  const unsigned asb0 = (unsigned)(uintptr_t)&As[0][0];
  const unsigned asb1 = (unsigned)(uintptr_t)&As[1][0];
  const unsigned aoff0 = (unsigned)(r0 * 4 + c0) * 16;
  const unsigned aoff1 = (unsigned)((64 + r0) * 4 + c0) * 16;

  v8f zero = {};
  v8f acc[2][4];
  #pragma unroll
  for (int s = 0; s < 2; ++s)
    #pragma unroll
    for (int t = 0; t < 4; ++t) acc[s][t] = zero;

  // prologue: stage tile 0 into buffer 0
  async_ld_b128(asb0 + aoff0, gA0);
  async_ld_b128(asb0 + aoff1, gA1);
  stage_B(W, F_DIM, 0, n0, &Bs[0][0], tid);
  wait_async0();
  __syncthreads();

  int cur = 0;
  for (int k0 = 0; k0 < D_DIM; k0 += 32) {
    const int nxt = cur ^ 1;
    if (k0 + 32 < D_DIM) {
      const unsigned asb = nxt ? asb1 : asb0;
      async_ld_b128(asb + aoff0, gA0 + k0 + 32);
      async_ld_b128(asb + aoff1, gA1 + k0 + 32);
      stage_B(W, F_DIM, k0 + 32, n0, &Bs[nxt][0], tid);
    }
    const __bf16* A = &As[cur][0];
    const __bf16* B = &Bs[cur][0];
    v16bf aF0 = load_frag_A(A, wm, lane);
    v16bf aF1 = load_frag_A(A, wm + 16, lane);
    #pragma unroll
    for (int t = 0; t < 4; ++t) {
      v16bf bF = load_frag_B(B, wn + 16 * t, lane);
      acc[0][t] = WMMA_BF16(aF0, bF, acc[0][t]);
      acc[1][t] = WMMA_BF16(aF1, bF, acc[1][t]);
    }
    wait_async0();
    __syncthreads();
    cur = nxt;
  }

  const float* b1e = b1 + (size_t)e * F_DIM;
  const int hi = lane >> 4, nl = lane & 15;
  #pragma unroll
  for (int s = 0; s < 2; ++s)
    #pragma unroll
    for (int t = 0; t < 4; ++t) {
      const int gn = n0 + wn + 16 * t + nl;
      const float bias = b1e[gn];
      #pragma unroll
      for (int j = 0; j < 8; ++j) {
        const int gm = m0 + wm + 16 * s + j + 8 * hi;
        float v = acc[s][t][j] + bias;
        h_buf[(size_t)gm * F_DIM + gn] = (__bf16)(v > 0.f ? v : 0.f);
      }
    }
}

// FFN2: out[token] += weight * (h[row, F] @ W2[e] + b2[e])
__global__ __launch_bounds__(256) void moe_ffn2(
    const __bf16* __restrict__ h_buf, const float* __restrict__ W2,
    const float* __restrict__ b2, const int* __restrict__ row_token,
    const float* __restrict__ row_weight, const int* __restrict__ padded_off,
    float* __restrict__ out) {
  __shared__ alignas(16) __bf16 As[2][128 * 32];
  __shared__ alignas(16) __bf16 Bs[2][128 * 32];
  const int m0 = blockIdx.y * 128;
  if (m0 >= padded_off[E_NUM]) return;
  int e = 0;
  while (e < E_NUM - 1 && m0 >= padded_off[e + 1]) ++e;
  const int n0 = blockIdx.x * 128;
  const float* W = W2 + (size_t)e * F_DIM * D_DIM;
  const int tid = threadIdx.x, lane = tid & 31, wid = tid >> 5;
  const int wm = (wid & 3) * 32, wn = (wid >> 2) * 64;
  const int r0 = tid >> 2, c0 = tid & 3;
  const __bf16* gA0 = h_buf + (size_t)(m0 + r0) * F_DIM + c0 * 8;
  const __bf16* gA1 = h_buf + (size_t)(m0 + 64 + r0) * F_DIM + c0 * 8;
  const unsigned asb0 = (unsigned)(uintptr_t)&As[0][0];
  const unsigned asb1 = (unsigned)(uintptr_t)&As[1][0];
  const unsigned aoff0 = (unsigned)(r0 * 4 + c0) * 16;
  const unsigned aoff1 = (unsigned)((64 + r0) * 4 + c0) * 16;

  v8f zero = {};
  v8f acc[2][4];
  #pragma unroll
  for (int s = 0; s < 2; ++s)
    #pragma unroll
    for (int t = 0; t < 4; ++t) acc[s][t] = zero;

  async_ld_b128(asb0 + aoff0, gA0);
  async_ld_b128(asb0 + aoff1, gA1);
  stage_B(W, D_DIM, 0, n0, &Bs[0][0], tid);
  wait_async0();
  __syncthreads();

  int cur = 0;
  for (int k0 = 0; k0 < F_DIM; k0 += 32) {
    const int nxt = cur ^ 1;
    if (k0 + 32 < F_DIM) {
      const unsigned asb = nxt ? asb1 : asb0;
      async_ld_b128(asb + aoff0, gA0 + k0 + 32);
      async_ld_b128(asb + aoff1, gA1 + k0 + 32);
      stage_B(W, D_DIM, k0 + 32, n0, &Bs[nxt][0], tid);
    }
    const __bf16* A = &As[cur][0];
    const __bf16* B = &Bs[cur][0];
    v16bf aF0 = load_frag_A(A, wm, lane);
    v16bf aF1 = load_frag_A(A, wm + 16, lane);
    #pragma unroll
    for (int t = 0; t < 4; ++t) {
      v16bf bF = load_frag_B(B, wn + 16 * t, lane);
      acc[0][t] = WMMA_BF16(aF0, bF, acc[0][t]);
      acc[1][t] = WMMA_BF16(aF1, bF, acc[1][t]);
    }
    wait_async0();
    __syncthreads();
    cur = nxt;
  }

  const float* b2e = b2 + (size_t)e * D_DIM;
  const int hi = lane >> 4, nl = lane & 15;
  #pragma unroll
  for (int s = 0; s < 2; ++s)
    #pragma unroll
    for (int t = 0; t < 4; ++t) {
      const int gn = n0 + wn + 16 * t + nl;
      const float bias = b2e[gn];
      #pragma unroll
      for (int j = 0; j < 8; ++j) {
        const int gm = m0 + wm + 16 * s + j + 8 * hi;
        const int tok = row_token[gm];
        const float wt = row_weight[gm];   // 0 for padding rows
        atomicAdd(out + (size_t)tok * D_DIM + gn, wt * (acc[s][t][j] + bias));
      }
    }
}

// ---------------- launch ----------------
extern "C" void kernel_launch(void* const* d_in, const int* in_sizes, int n_in,
                              void* d_out, int out_size, void* d_ws, size_t ws_size,
                              hipStream_t stream) {
  (void)in_sizes; (void)n_in; (void)out_size; (void)ws_size;
  const float* x  = (const float*)d_in[0];
  const float* Wg = (const float*)d_in[1];
  const float* bg = (const float*)d_in[2];
  const float* W1 = (const float*)d_in[3];
  const float* b1 = (const float*)d_in[4];
  const float* W2 = (const float*)d_in[5];
  const float* b2 = (const float*)d_in[6];
  float* out = (float*)d_out;

  uint8_t* ws = (uint8_t*)d_ws;
  size_t off = 0;
  auto alloc = [&](size_t bytes) -> void* {
    off = (off + 255) & ~(size_t)255;
    void* p = ws + off;
    off += bytes;
    return p;
  };
  int*    pair_count = (int*)   alloc(E_NUM * 4);
  int*    cursor     = (int*)   alloc(E_NUM * 4);
  int*    padded_off = (int*)   alloc((E_NUM + 1) * 4);
  float*  top1_cnt   = (float*) alloc(E_NUM * 4);
  float*  prob_sums  = (float*) alloc(E_NUM * 4);
  int*    tok_e      = (int*)   alloc((size_t)2 * T_TOK * 4);
  float*  tok_p      = (float*) alloc((size_t)2 * T_TOK * 4);
  int*    row_token  = (int*)   alloc((size_t)ROWS_CAP * 4);
  float*  row_weight = (float*) alloc((size_t)ROWS_CAP * 4);
  __bf16* xb         = (__bf16*)alloc((size_t)T_TOK * D_DIM * 2);
  __bf16* h_buf      = (__bf16*)alloc((size_t)ROWS_CAP * F_DIM * 2);

  hipLaunchKernelGGL(zero_f32, dim3((T_TOK * D_DIM + 255) / 256), dim3(256), 0, stream,
                     out, T_TOK * D_DIM);
  hipLaunchKernelGGL(init_route, dim3((ROWS_CAP + 255) / 256), dim3(256), 0, stream,
                     pair_count, cursor, top1_cnt, prob_sums, row_token, row_weight);
  hipLaunchKernelGGL(gate_kernel, dim3(T_TOK / 256), dim3(256), 0, stream,
                     x, Wg, bg, pair_count, top1_cnt, prob_sums, tok_e, tok_p);
  hipLaunchKernelGGL(offsets_kernel, dim3(1), dim3(1), 0, stream,
                     pair_count, top1_cnt, prob_sums, padded_off, out + (size_t)T_TOK * D_DIM);
  hipLaunchKernelGGL(scatter_kernel, dim3(T_TOK / 256), dim3(256), 0, stream,
                     tok_e, tok_p, padded_off, cursor, row_token, row_weight);
  hipLaunchKernelGGL(cvt_bf16, dim3((T_TOK * D_DIM / 4) / 256), dim3(256), 0, stream,
                     x, xb, T_TOK * D_DIM / 4);
  hipLaunchKernelGGL(moe_ffn1, dim3(F_DIM / 128, ROW_TILES), dim3(256), 0, stream,
                     xb, W1, b1, row_token, padded_off, h_buf);
  hipLaunchKernelGGL(moe_ffn2, dim3(D_DIM / 128, ROW_TILES), dim3(256), 0, stream,
                     h_buf, W2, b2, row_token, row_weight, padded_off, out);
}
